// QuantAttention_68341519614193
// MI455X (gfx1250) — matrix operand order repaired and checked by reading, not compiled, not run on hardware
//
#include <hip/hip_runtime.h>
#include <hip/hip_bf16.h>

typedef __bf16 bf16t;
typedef __attribute__((ext_vector_type(16))) __bf16 v16bf;
typedef __attribute__((ext_vector_type(8)))  float  v8f;

#define NH  12
#define HD  64
#define SEQ 4096
#define CCH 768

#define WMMA(a,b,c) __builtin_amdgcn_wmma_f32_16x16x32_bf16(false,(a),false,(b),(short)0,(c),false,false)

static __device__ inline v8f v8f_zero() {
  v8f z = {0.f,0.f,0.f,0.f,0.f,0.f,0.f,0.f};
  return z;
}

// ---- CDNA5 async global->LDS copy (ASYNCcnt-tracked), 16 bytes per op ------
// VDST VGPR = wave-relative LDS byte offset = low 32 bits of the generic
// pointer to a __shared__ object; VADDR = 64-bit global address pair.
static __device__ inline void async_copy_b128(const bf16t* __restrict__ gsrc,
                                              bf16t* lds_dst) {
  const unsigned lds_off = (unsigned)(reinterpret_cast<uintptr_t>(lds_dst));
  asm volatile("global_load_async_to_lds_b128 %0, %1, off"
               :: "v"(lds_off), "v"(gsrc) : "memory");
}
#define WAIT_ASYNC(n) asm volatile("s_wait_asynccnt %0" :: "i"(n) : "memory")

// A-matrix fragment (16x32 bf16, MxK), row-major source.
// lanes 0-15 row M=lane, K=0..15; lanes 16-31 K=16..31 (contiguous, matches B).
static __device__ inline v16bf fragA_bf16(const bf16t* __restrict__ row) {
  const int lane = threadIdx.x & 31;
  const int kb = (lane < 16) ? 0 : 16;
  union { uint4 u[2]; v16bf v; } c;
  c.u[0] = *reinterpret_cast<const uint4*>(row + kb);
  c.u[1] = *reinterpret_cast<const uint4*>(row + kb + 8);
  return c.v;
}

// B-matrix fragment (32x16 bf16, KxN); `col` = this lane's column, contiguous over K.
// Same contiguous K order as fragA: lanes 0-15 K=0..15, lanes 16-31 K=16..31 (sequential).
static __device__ inline v16bf fragB_bf16(const bf16t* __restrict__ col) {
  const int lane = threadIdx.x & 31;
  const int kb = (lane < 16) ? 0 : 16;
  union { uint4 u[2]; v16bf v; } c;
  c.u[0] = *reinterpret_cast<const uint4*>(col + kb);
  c.u[1] = *reinterpret_cast<const uint4*>(col + kb + 8);
  return c.v;
}

static __device__ inline v16bf fragB_f32(const float* __restrict__ col) {
  const int lane = threadIdx.x & 31;
  const int kb = (lane < 16) ? 0 : 16;
  v16bf f;
#pragma unroll
  for (int i = 0; i < 16; ++i) f[i] = (bf16t)col[kb + i];
  return f;
}

// Stage one 128x32 bf16 tile (row-major, src_ld elements/row) into LDS via
// async copies: 512 x b128; 256 threads -> 2 ops/thread.
static __device__ inline void stage_tile_async(const bf16t* __restrict__ src,
                                               int src_ld, bf16t* lds) {
  const int t = threadIdx.x;
#pragma unroll
  for (int h = 0; h < 2; ++h) {
    const int idx = t + h * 256;          // 0..511
    const int row = idx >> 2, seg = (idx & 3) * 8;
    async_copy_b128(src + row * src_ld + seg, lds + row * 32 + seg);
  }
}

// ---------------------------------------------------------------------------
// Kernel 0a/0b: fp32 -> bf16 pre-conversion (and transpose for weights) so
// the GEMM main loops can use pure async global->LDS staging.
// ---------------------------------------------------------------------------
__global__ __launch_bounds__(256) void k_cvt(const float* __restrict__ src,
                                             bf16t* __restrict__ dst, int n) {
  const int i = blockIdx.x * 256 + threadIdx.x;
  if (i < n) dst[i] = (bf16t)src[i];
}

// src is (R x C) row-major; dst = src^T as (C x R) row-major.
__global__ __launch_bounds__(256) void k_cvtT(const float* __restrict__ src,
                                              bf16t* __restrict__ dst,
                                              int R, int C) {
  const int i = blockIdx.x * 256 + threadIdx.x;
  if (i < R * C) {
    const int c = i / R, r = i - c * R;
    dst[i] = (bf16t)src[r * C + c];
  }
}

// ---------------------------------------------------------------------------
// Kernel 1: QKV projection.  xb(4096x768 bf16) @ wqkvT^T + b -> q,k row-major
// bf16 [head][m][d]; v transposed bf16 [head][d][m].  q stored UNscaled.
// Double-buffered async-staged 128x128x32 tiles, 8 waves, 8 WMMA/wave/step.
// ---------------------------------------------------------------------------
__global__ __launch_bounds__(256) void k_qkv(const bf16t* __restrict__ X,
                                             const bf16t* __restrict__ WT,  // (2304 x 768)
                                             const float* __restrict__ B,
                                             bf16t* __restrict__ q,
                                             bf16t* __restrict__ kk,
                                             bf16t* __restrict__ vT) {
  __shared__ bf16t As[2 * 128 * 32];
  __shared__ bf16t Bs[2 * 128 * 32];
  const int tid  = threadIdx.x;
  const int lane = tid & 31;
  const int wv   = tid >> 5;
  const int wy = wv >> 2, wx = wv & 3;          // 2x4 wave grid -> 64x32 per wave
  const int m0 = blockIdx.y * 128, n0 = blockIdx.x * 128;
  const int NT = CCH / 32;                      // 24 K-tiles

  v8f acc[4][2];
#pragma unroll
  for (int i = 0; i < 4; ++i)
#pragma unroll
    for (int j = 0; j < 2; ++j) acc[i][j] = v8f_zero();

  const bf16t* Abase = X  + (long)m0 * CCH;
  const bf16t* Bbase = WT + (long)n0 * CCH;

  int cur = 0;
  stage_tile_async(Abase, CCH, As);
  stage_tile_async(Bbase, CCH, Bs);

  for (int kt = 0; kt < NT; ++kt) {
    const int nxt = cur ^ 1;
    if (kt + 1 < NT) {
      const int k1 = (kt + 1) * 32;
      stage_tile_async(Abase + k1, CCH, As + nxt * 4096);
      stage_tile_async(Bbase + k1, CCH, Bs + nxt * 4096);
      WAIT_ASYNC(4);          // in-order: current tile's 4 ops/thread are done
    } else {
      WAIT_ASYNC(0);
    }
    __syncthreads();

    const bf16t* Ac = As + cur * 4096;
    const bf16t* Bc = Bs + cur * 4096;
    v16bf a[4], b[2];
#pragma unroll
    for (int i = 0; i < 4; ++i)
      a[i] = fragA_bf16(Ac + (wy * 64 + 16 * i + (lane & 15)) * 32);
#pragma unroll
    for (int j = 0; j < 2; ++j)
      b[j] = fragB_bf16(Bc + (wx * 32 + 16 * j + (lane & 15)) * 32);
#pragma unroll
    for (int i = 0; i < 4; ++i)
#pragma unroll
      for (int j = 0; j < 2; ++j) acc[i][j] = WMMA(a[i], b[j], acc[i][j]);

    __syncthreads();          // everyone done reading `cur` before re-staging it
    cur = nxt;
  }

  const int rowoff = (lane < 16) ? 0 : 8;
#pragma unroll
  for (int i = 0; i < 4; ++i)
#pragma unroll
    for (int j = 0; j < 2; ++j)
#pragma unroll
      for (int v = 0; v < 8; ++v) {
        const int m = m0 + wy * 64 + 16 * i + v + rowoff;
        const int n = n0 + wx * 32 + 16 * j + (lane & 15);
        const float c = acc[i][j][v] + B[n];
        const int which = n / CCH;
        const int rem = n - which * CCH;
        const int head = rem >> 6, d = rem & 63;
        if (which == 0)      q [(head * SEQ + m) * HD + d] = (bf16t)c;
        else if (which == 1) kk[(head * SEQ + m) * HD + d] = (bf16t)c;
        else                 vT[(head * HD + d) * SEQ + m] = (bf16t)c;
      }
}

// ---------------------------------------------------------------------------
// Kernel 2: decomposed rel-pos bias tables.
//   mode 0: relH[head][h*64+w][j] = sum_c q[head][h*64+w][c] * rel_pos_h[h-j+63][c]
//   mode 1: relW[head][h*64+w][j] = sum_c q[head][h*64+w][c] * rel_pos_w[w-j+63][c]
// One block per (i, head, mode); 4 waves x 16 rows; M=N=K=64.
// ---------------------------------------------------------------------------
__global__ __launch_bounds__(128) void k_rel(const bf16t* __restrict__ q,
                                             const float* __restrict__ rel_pos_h,
                                             const float* __restrict__ rel_pos_w,
                                             float* __restrict__ relH,
                                             float* __restrict__ relW) {
  const int i    = blockIdx.x;     // fixed h (mode 0) or fixed w (mode 1)
  const int head = blockIdx.y;
  const int mode = blockIdx.z;
  const int lane = threadIdx.x & 31;
  const int wv   = threadIdx.x >> 5;

  const float* rp = (mode == 0) ? rel_pos_h : rel_pos_w;
  float* outT     = (mode == 0) ? relH : relW;

  const int t = wv * 16 + (lane & 15);
  const long rowstride = (mode == 0) ? (long)HD : (long)64 * HD;
  const bf16t* abase = q + (long)head * SEQ * HD
                         + ((mode == 0) ? (long)i * 64 * HD : (long)i * HD)
                         + (long)t * rowstride;

  v8f acc[4];
#pragma unroll
  for (int jt = 0; jt < 4; ++jt) acc[jt] = v8f_zero();

#pragma unroll
  for (int ks = 0; ks < 2; ++ks) {
    v16bf a = fragA_bf16(abase + ks * 32);
#pragma unroll
    for (int jt = 0; jt < 4; ++jt) {
      const int j = jt * 16 + (lane & 15);
      const float* brow = rp + (i - j + 63) * HD + ks * 32;  // index in [0,126]
      v16bf b = fragB_f32(brow);
      acc[jt] = WMMA(a, b, acc[jt]);
    }
  }

  const int rowoff = (lane < 16) ? 0 : 8;
#pragma unroll
  for (int jt = 0; jt < 4; ++jt)
#pragma unroll
    for (int v = 0; v < 8; ++v) {
      const int tt = wv * 16 + v + rowoff;
      const int m = (mode == 0) ? (i * 64 + tt) : (tt * 64 + i);
      const int j = jt * 16 + (lane & 15);
      outT[((long)head * SEQ + m) * HD + j] = acc[jt][v];
    }
}

// ---------------------------------------------------------------------------
// Kernel 3: flash attention.  Per wave: 16 query rows, sweep 4096 keys in
// 32-wide steps.  S = 0.125*(q.kT) + relH + relW, online softmax, O += P@V.
// Block = 8 waves = 128 query rows; grid = (32 m-blocks, 12 heads).
// ---------------------------------------------------------------------------
__global__ __launch_bounds__(256) void k_flash(const bf16t* __restrict__ q,
                                               const bf16t* __restrict__ kmat,
                                               const bf16t* __restrict__ vT,
                                               const float* __restrict__ relH,
                                               const float* __restrict__ relW,
                                               bf16t* __restrict__ o) {
  __shared__ bf16t Pl[8 * 16 * 32];        // per-wave P staging tile (16x32)
  const int head = blockIdx.y;
  const int m0   = blockIdx.x * 128;
  const int lane = threadIdx.x & 31;
  const int wv   = threadIdx.x >> 5;
  const int mbase  = m0 + wv * 16;
  const int rowoff = (lane < 16) ? 0 : 8;
  const float scale = 0.125f;

  const bf16t* qb = q + ((long)head * SEQ + mbase + (lane & 15)) * HD;
  const v16bf aq0 = fragA_bf16(qb);
  const v16bf aq1 = fragA_bf16(qb + 32);

  const float* rH = relH + ((long)head * SEQ + mbase) * HD;
  const float* rW = relW + ((long)head * SEQ + mbase) * HD;

  v8f acc_o[4];
#pragma unroll
  for (int jd = 0; jd < 4; ++jd) acc_o[jd] = v8f_zero();
  float mrun[8], lrun[8];
#pragma unroll
  for (int v = 0; v < 8; ++v) { mrun[v] = -3.0e38f; lrun[v] = 0.f; }

  bf16t* pl = Pl + wv * (16 * 32);

  for (int n0 = 0; n0 < SEQ; n0 += 32) {
    // ---- S tile: 16x32, 4 WMMAs ----
    v8f s[2];
#pragma unroll
    for (int jn = 0; jn < 2; ++jn) {
      s[jn] = v8f_zero();
      const bf16t* kcol = kmat + ((long)head * SEQ + n0 + jn * 16 + (lane & 15)) * HD;
      s[jn] = WMMA(aq0, fragB_bf16(kcol),      s[jn]);
      s[jn] = WMMA(aq1, fragB_bf16(kcol + 32), s[jn]);
    }

    // ---- scale + bias + online softmax ----
    const int hp = n0 >> 6;          // key h' (constant over this 32-step)
    const int wbase = n0 & 63;       // key w' base
    float sv[2][8], mx[8];
#pragma unroll
    for (int v = 0; v < 8; ++v) {
      const int r = v + rowoff;
      const float bh = rH[r * HD + hp];
      const float e0 = s[0][v] * scale + bh + rW[r * HD + wbase +      (lane & 15)];
      const float e1 = s[1][v] * scale + bh + rW[r * HD + wbase + 16 + (lane & 15)];
      sv[0][v] = e0; sv[1][v] = e1;
      mx[v] = fmaxf(e0, e1);
    }
#pragma unroll
    for (int off = 1; off < 16; off <<= 1)
#pragma unroll
      for (int v = 0; v < 8; ++v) mx[v] = fmaxf(mx[v], __shfl_xor(mx[v], off, 32));

    float fac[8], ps[8];
#pragma unroll
    for (int v = 0; v < 8; ++v) {
      const float nm = fmaxf(mrun[v], mx[v]);
      fac[v] = __expf(mrun[v] - nm);
      mrun[v] = nm;
      const float p0 = __expf(sv[0][v] - nm);
      const float p1 = __expf(sv[1][v] - nm);
      sv[0][v] = p0; sv[1][v] = p1;
      ps[v] = p0 + p1;
    }
#pragma unroll
    for (int off = 1; off < 16; off <<= 1)
#pragma unroll
      for (int v = 0; v < 8; ++v) ps[v] += __shfl_xor(ps[v], off, 32);
#pragma unroll
    for (int v = 0; v < 8; ++v) lrun[v] = lrun[v] * fac[v] + ps[v];
#pragma unroll
    for (int jd = 0; jd < 4; ++jd)
#pragma unroll
      for (int v = 0; v < 8; ++v) acc_o[jd][v] *= fac[v];

    // ---- re-layout P (C-layout -> A-layout) via per-wave LDS tile ----
#pragma unroll
    for (int jn = 0; jn < 2; ++jn)
#pragma unroll
      for (int v = 0; v < 8; ++v)
        pl[(v + rowoff) * 32 + jn * 16 + (lane & 15)] = (bf16t)sv[jn][v];
    asm volatile("s_wait_dscnt 0" ::: "memory");   // LDS in-order per wave
    const v16bf pa = fragA_bf16(pl + (lane & 15) * 32);

    // ---- O += P @ V : 4 WMMAs ----
#pragma unroll
    for (int jd = 0; jd < 4; ++jd) {
      const bf16t* vcol = vT + ((long)head * HD + jd * 16 + (lane & 15)) * SEQ + n0;
      acc_o[jd] = WMMA(pa, fragB_bf16(vcol), acc_o[jd]);
    }
  }

  float inv[8];
#pragma unroll
  for (int v = 0; v < 8; ++v) inv[v] = 1.0f / lrun[v];
#pragma unroll
  for (int jd = 0; jd < 4; ++jd)
#pragma unroll
    for (int v = 0; v < 8; ++v) {
      const int m = mbase + v + rowoff;
      o[(long)m * CCH + head * HD + jd * 16 + (lane & 15)] = (bf16t)(acc_o[jd][v] * inv[v]);
    }
}

// ---------------------------------------------------------------------------
// Kernel 4: output projection.  o(4096x768 bf16) @ woT^T + b_o -> f32.
// Same async double-buffered structure as k_qkv.
// ---------------------------------------------------------------------------
__global__ __launch_bounds__(256) void k_oproj(const bf16t* __restrict__ A,
                                               const bf16t* __restrict__ WT,  // (768 x 768)
                                               const float* __restrict__ B,
                                               float* __restrict__ out) {
  __shared__ bf16t As[2 * 128 * 32];
  __shared__ bf16t Bs[2 * 128 * 32];
  const int tid  = threadIdx.x;
  const int lane = tid & 31;
  const int wv   = tid >> 5;
  const int wy = wv >> 2, wx = wv & 3;
  const int m0 = blockIdx.y * 128, n0 = blockIdx.x * 128;
  const int NT = CCH / 32;

  v8f acc[4][2];
#pragma unroll
  for (int i = 0; i < 4; ++i)
#pragma unroll
    for (int j = 0; j < 2; ++j) acc[i][j] = v8f_zero();

  const bf16t* Abase = A  + (long)m0 * CCH;
  const bf16t* Bbase = WT + (long)n0 * CCH;

  int cur = 0;
  stage_tile_async(Abase, CCH, As);
  stage_tile_async(Bbase, CCH, Bs);

  for (int kt = 0; kt < NT; ++kt) {
    const int nxt = cur ^ 1;
    if (kt + 1 < NT) {
      const int k1 = (kt + 1) * 32;
      stage_tile_async(Abase + k1, CCH, As + nxt * 4096);
      stage_tile_async(Bbase + k1, CCH, Bs + nxt * 4096);
      WAIT_ASYNC(4);
    } else {
      WAIT_ASYNC(0);
    }
    __syncthreads();

    const bf16t* Ac = As + cur * 4096;
    const bf16t* Bc = Bs + cur * 4096;
    v16bf a[4], b[2];
#pragma unroll
    for (int i = 0; i < 4; ++i)
      a[i] = fragA_bf16(Ac + (wy * 64 + 16 * i + (lane & 15)) * 32);
#pragma unroll
    for (int j = 0; j < 2; ++j)
      b[j] = fragB_bf16(Bc + (wx * 32 + 16 * j + (lane & 15)) * 32);
#pragma unroll
    for (int i = 0; i < 4; ++i)
#pragma unroll
      for (int j = 0; j < 2; ++j) acc[i][j] = WMMA(a[i], b[j], acc[i][j]);

    __syncthreads();
    cur = nxt;
  }

  const int rowoff = (lane < 16) ? 0 : 8;
#pragma unroll
  for (int i = 0; i < 4; ++i)
#pragma unroll
    for (int j = 0; j < 2; ++j)
#pragma unroll
      for (int v = 0; v < 8; ++v) {
        const int m = m0 + wy * 64 + 16 * i + v + rowoff;
        const int n = n0 + wx * 32 + 16 * j + (lane & 15);
        out[(long)m * CCH + n] = acc[i][j][v] + B[n];
      }
}

// ---------------------------------------------------------------------------
extern "C" void kernel_launch(void* const* d_in, const int* in_sizes, int n_in,
                              void* d_out, int out_size, void* d_ws, size_t ws_size,
                              hipStream_t stream) {
  (void)in_sizes; (void)n_in; (void)out_size; (void)ws_size;
  const float* x         = (const float*)d_in[0];
  const float* w_qkv     = (const float*)d_in[1];
  const float* b_qkv     = (const float*)d_in[2];
  const float* w_o       = (const float*)d_in[3];
  const float* b_o       = (const float*)d_in[4];
  const float* rel_pos_h = (const float*)d_in[5];
  const float* rel_pos_w = (const float*)d_in[6];

  char* ws = (char*)d_ws;
  const size_t HEADBUF = (size_t)NH * SEQ * HD * sizeof(bf16t);   // 6.29 MB
  const size_t RELBUF  = (size_t)NH * SEQ * HD * sizeof(float);   // 12.58 MB
  const size_t XBUF    = (size_t)SEQ * CCH * sizeof(bf16t);       // 6.29 MB
  const size_t WQKVBUF = (size_t)3 * CCH * CCH * sizeof(bf16t);   // 3.54 MB
  const size_t WOBUF   = (size_t)CCH * CCH * sizeof(bf16t);       // 1.18 MB

  size_t off = 0;
  bf16t* q_ws   = (bf16t*)(ws + off); off += HEADBUF;
  bf16t* k_ws   = (bf16t*)(ws + off); off += HEADBUF;
  bf16t* vT_ws  = (bf16t*)(ws + off); off += HEADBUF;
  float* relH   = (float*)(ws + off); off += RELBUF;
  float* relW   = (float*)(ws + off); off += RELBUF;
  bf16t* o_ws   = (bf16t*)(ws + off); off += XBUF;
  bf16t* xb     = (bf16t*)(ws + off); off += XBUF;
  bf16t* wqkvT  = (bf16t*)(ws + off); off += WQKVBUF;
  bf16t* woT    = (bf16t*)(ws + off); off += WOBUF;

  // 0) fp32 -> bf16 pre-conversion (+ weight transpose to n-major)
  k_cvt  <<<dim3((SEQ * CCH + 255) / 256),        256, 0, stream>>>(x, xb, SEQ * CCH);
  k_cvtT <<<dim3((3 * CCH * CCH + 255) / 256),    256, 0, stream>>>(w_qkv, wqkvT, CCH, 3 * CCH);
  k_cvtT <<<dim3((CCH * CCH + 255) / 256),        256, 0, stream>>>(w_o, woT, CCH, CCH);

  // 1..4) WMMA pipeline
  k_qkv  <<<dim3(18, 32),    256, 0, stream>>>(xb, wqkvT, b_qkv, q_ws, k_ws, vT_ws);
  k_rel  <<<dim3(64, NH, 2), 128, 0, stream>>>(q_ws, rel_pos_h, rel_pos_w, relH, relW);
  k_flash<<<dim3(32, NH),    256, 0, stream>>>(q_ws, k_ws, vT_ws, relH, relW, o_ws);
  k_oproj<<<dim3(6, 32),     256, 0, stream>>>(o_ws, woT, b_o, (float*)d_out);
}
